// CosineEmbeddingLoss_11682311045882
// MI455X (gfx1250) — matrix operand order repaired and checked by reading, not compiled, not run on hardware
//
#include <hip/hip_runtime.h>

#define NROWS 8192
#define DIM   256
#define BT    64      // block output tile (BT x BT)
#define KC    64      // K chunk staged in LDS (two K=32 WMMA groups)
#define NCHUNK (DIM / KC)
#define LSTR  72      // LDS row stride in bf16 (KC + 8 pad -> bank-striped frags)
#define MARGIN 0.5f
#define EPS    1e-8f

typedef float  v8f   __attribute__((ext_vector_type(8)));
typedef __bf16 v4bf  __attribute__((ext_vector_type(4)));
typedef __bf16 v8bf  __attribute__((ext_vector_type(8)));
typedef __bf16 v16bf __attribute__((ext_vector_type(16)));

// ---------------------------------------------------------------------------
// Kernel 1: row-normalize (clamped L2 norm) and split fp32 -> bf16 hi + lo.
// One wave32 per row.
// ---------------------------------------------------------------------------
__global__ __launch_bounds__(256) void normalize_split(
    const float* __restrict__ in,
    __bf16* __restrict__ hiE, __bf16* __restrict__ loE)
{
    const int row  = blockIdx.x * 8 + (threadIdx.x >> 5);
    const int lane = threadIdx.x & 31;
    const float4* rp = (const float4*)(in + (size_t)row * DIM);

    float4 v0 = rp[lane];
    float4 v1 = rp[lane + 32];
    float ss = v0.x*v0.x + v0.y*v0.y + v0.z*v0.z + v0.w*v0.w
             + v1.x*v1.x + v1.y*v1.y + v1.z*v1.z + v1.w*v1.w;
    #pragma unroll
    for (int off = 16; off > 0; off >>= 1)
        ss += __shfl_xor(ss, off, 32);

    const float inv = 1.0f / fmaxf(sqrtf(ss), EPS);

    float x[8] = { v0.x*inv, v0.y*inv, v0.z*inv, v0.w*inv,
                   v1.x*inv, v1.y*inv, v1.z*inv, v1.w*inv };
    __bf16 h[8], l[8];
    #pragma unroll
    for (int i = 0; i < 8; ++i) {
        h[i] = (__bf16)x[i];
        l[i] = (__bf16)(x[i] - (float)h[i]);   // residual -> ~16-bit mantissa total
    }

    const size_t base = (size_t)row * DIM + lane * 4;
    *(v4bf*)(hiE + base)       = v4bf{h[0], h[1], h[2], h[3]};
    *(v4bf*)(hiE + base + 128) = v4bf{h[4], h[5], h[6], h[7]};
    *(v4bf*)(loE + base)       = v4bf{l[0], l[1], l[2], l[3]};
    *(v4bf*)(loE + base + 128) = v4bf{l[4], l[5], l[6], l[7]};
}

// ---------------------------------------------------------------------------
// Kernel 2: fused sim-GEMM via split-bf16 (hi*hi + hi*lo + lo*hi, fp32 acc)
// using v_wmma_f32_16x16x32_bf16, double-buffered LDS fed by
// GLOBAL_LOAD_ASYNC_TO_LDS_B128 (ASYNCcnt), + masked loss epilogue.
// Grid (128,128); only bj >= bi tiles computed (weight 2 off-diagonal).
// ---------------------------------------------------------------------------
__global__ __launch_bounds__(128) void gemm_loss(
    const __bf16* __restrict__ Ehi, const __bf16* __restrict__ Elo,
    const int* __restrict__ labels, float* __restrict__ partials)
{
    __shared__ __bf16 AsH[2][BT][LSTR];
    __shared__ __bf16 AsL[2][BT][LSTR];
    __shared__ __bf16 BsH[2][BT][LSTR];
    __shared__ __bf16 BsL[2][BT][LSTR];
    __shared__ float  red[128];

    const int bj  = blockIdx.x;
    const int bi  = blockIdx.y;
    const int pid = bi * gridDim.x + bj;

    if (bj < bi) {                       // lower triangle: contribute zero
        if (threadIdx.x == 0) partials[pid] = 0.0f;
        return;
    }

    const int tid  = threadIdx.x;        // 0..127
    const int wave = tid >> 5;           // 0..3 (wave32)
    const int lane = tid & 31;
    const int row0 = bi * BT;
    const int col0 = bj * BT;

    v8f acc[4];
    #pragma unroll
    for (int s = 0; s < 4; ++s) acc[s] = (v8f)0.0f;

    const int m    = lane & 15;
    const int half = lane >> 4;
    const int arow = wave * 16 + m;      // wave's A-strip row within tile

    // Per-thread staging geometry: 16 async b128 per chunk (4 rows x 4 tiles).
    const int r0 = tid >> 3;             // base row (stride 16 over i)
    const int c8 = (tid & 7) * 8;        // bf16 column within row

    // Issue one KC-chunk of async global->LDS copies into buffer b.
    auto issue_chunk = [&](int k0, int b) {
        #pragma unroll
        for (int i = 0; i < 4; ++i) {
            const int r = 16 * i + r0;
            const unsigned ga = (unsigned)(((row0 + r) * DIM + k0 + c8) * 2);
            const unsigned gb = (unsigned)(((col0 + r) * DIM + k0 + c8) * 2);
            // flat->LDS: hardware truncates to addr[31:0]
            const unsigned lah = (unsigned)(uintptr_t)&AsH[b][r][c8];
            const unsigned lal = (unsigned)(uintptr_t)&AsL[b][r][c8];
            const unsigned lbh = (unsigned)(uintptr_t)&BsH[b][r][c8];
            const unsigned lbl = (unsigned)(uintptr_t)&BsL[b][r][c8];
            asm volatile("global_load_async_to_lds_b128 %0, %1, %2"
                         :: "v"(lah), "v"(ga), "s"(Ehi) : "memory");
            asm volatile("global_load_async_to_lds_b128 %0, %1, %2"
                         :: "v"(lal), "v"(ga), "s"(Elo) : "memory");
            asm volatile("global_load_async_to_lds_b128 %0, %1, %2"
                         :: "v"(lbh), "v"(gb), "s"(Ehi) : "memory");
            asm volatile("global_load_async_to_lds_b128 %0, %1, %2"
                         :: "v"(lbl), "v"(gb), "s"(Elo) : "memory");
        }
    };

    issue_chunk(0, 0);                   // prologue: stage chunk 0

    for (int c = 0; c < NCHUNK; ++c) {
        const int cb = c & 1;
        if (c + 1 < NCHUNK) {
            issue_chunk((c + 1) * KC, (c + 1) & 1);  // prefetch next chunk
            // 16 newest (next chunk) may remain in flight; chunk c is done.
            asm volatile("s_wait_asynccnt 0x10" ::: "memory");
        } else {
            asm volatile("s_wait_asynccnt 0x0" ::: "memory");
        }
        __syncthreads();                 // all waves' copies visible

        #pragma unroll
        for (int g = 0; g < 2; ++g) {            // two K=32 groups per chunk
            const int kb = g * 32;
            // A frag: contiguous layout matching B: elems 0..7 = K 16h..16h+7,
            //         elems 8..15 = K 16h+8..16h+15
            v8bf a0h = *(const v8bf*)&AsH[cb][arow][kb + 16 * half];
            v8bf a1h = *(const v8bf*)&AsH[cb][arow][kb + 16 * half + 8];
            v8bf a0l = *(const v8bf*)&AsL[cb][arow][kb + 16 * half];
            v8bf a1l = *(const v8bf*)&AsL[cb][arow][kb + 16 * half + 8];
            v16bf ahi = __builtin_shufflevector(a0h, a1h,
                0,1,2,3,4,5,6,7,8,9,10,11,12,13,14,15);
            v16bf alo = __builtin_shufflevector(a0l, a1l,
                0,1,2,3,4,5,6,7,8,9,10,11,12,13,14,15);

            #pragma unroll
            for (int s = 0; s < 4; ++s) {
                const int bcol = s * 16 + m;
                // B frag: 32x16 bf16 layout: elems 0..15 = K 16h..16h+15
                v16bf bhi = *(const v16bf*)&BsH[cb][bcol][kb + 16 * half];
                v16bf blo = *(const v16bf*)&BsL[cb][bcol][kb + 16 * half];

                acc[s] = __builtin_amdgcn_wmma_f32_16x16x32_bf16(
                    false, ahi, false, bhi, (short)0, acc[s], false, false);
                acc[s] = __builtin_amdgcn_wmma_f32_16x16x32_bf16(
                    false, ahi, false, blo, (short)0, acc[s], false, false);
                acc[s] = __builtin_amdgcn_wmma_f32_16x16x32_bf16(
                    false, alo, false, bhi, (short)0, acc[s], false, false);
            }
        }
        __syncthreads();                 // done reading before next overwrite
    }

    // ---- epilogue: masked contrastive loss on in-register sim tile ----
    const int gibase = row0 + wave * 16 + half * 8;  // C: VGPR r -> M = r + 8*half

    int li[8];
    #pragma unroll
    for (int r = 0; r < 8; ++r) li[r] = labels[gibase + r];

    float lsum = 0.0f;
    #pragma unroll
    for (int s = 0; s < 4; ++s) {
        const int gj = col0 + s * 16 + m;
        const int lj = labels[gj];
        #pragma unroll
        for (int r = 0; r < 8; ++r) {
            const float sim = acc[s][r];
            const int   gi  = gibase + r;
            float v;
            if (gi == gj)          v = 0.0f;                   // off-diag mask
            else if (li[r] == lj)  v = 1.0f - sim;             // positive pair
            else                   v = fmaxf(sim - MARGIN, 0.0f); // negative
            lsum += v;
        }
    }
    lsum *= (bi == bj) ? 1.0f : 2.0f;   // symmetry weight

    // deterministic block reduction -> one partial per block
    red[tid] = lsum;
    __syncthreads();
    #pragma unroll
    for (int st = 64; st > 0; st >>= 1) {
        if (tid < st) red[tid] += red[tid + st];
        __syncthreads();
    }
    if (tid == 0) partials[pid] = red[0];
}

// ---------------------------------------------------------------------------
// Kernel 3: final reduction -> mean over N*N entries.
// ---------------------------------------------------------------------------
__global__ __launch_bounds__(256) void reduce_partials(
    const float* __restrict__ partials, int n, float* __restrict__ out)
{
    __shared__ float s[256];
    float v = 0.0f;
    for (int i = threadIdx.x; i < n; i += 256) v += partials[i];
    s[threadIdx.x] = v;
    __syncthreads();
    #pragma unroll
    for (int st = 128; st > 0; st >>= 1) {
        if (threadIdx.x < st) s[threadIdx.x] += s[threadIdx.x + st];
        __syncthreads();
    }
    if (threadIdx.x == 0)
        out[0] = s[0] * (1.0f / ((float)NROWS * (float)NROWS));
}

// ---------------------------------------------------------------------------
extern "C" void kernel_launch(void* const* d_in, const int* in_sizes, int n_in,
                              void* d_out, int out_size, void* d_ws, size_t ws_size,
                              hipStream_t stream)
{
    const float* emb    = (const float*)d_in[0];
    const int*   labels = (const int*)d_in[1];
    float*       out    = (float*)d_out;

    __bf16* Ehi = (__bf16*)d_ws;                          // 8192*256 bf16 (4 MB)
    __bf16* Elo = Ehi + (size_t)NROWS * DIM;              // 4 MB
    float*  partials = (float*)(Elo + (size_t)NROWS * DIM); // 128*128 floats

    normalize_split<<<NROWS / 8, 256, 0, stream>>>(emb, Ehi, Elo);

    dim3 grid(NROWS / BT, NROWS / BT);                    // 128 x 128 tiles
    gemm_loss<<<grid, 128, 0, stream>>>(Ehi, Elo, labels, partials);

    reduce_partials<<<1, 256, 0, stream>>>(partials, (NROWS / BT) * (NROWS / BT), out);
}
